// GraphAttentionNetwork_64192581206329
// MI455X (gfx1250) — matrix operand order repaired and checked by reading, not compiled, not run on hardware
//
#include <hip/hip_runtime.h>
#include <math.h>

// GAT dims (from reference)
#define HEADS 4
#define HIDC  128
#define FIN   256
#define FOUT  64
#define H1W   (HEADS * HIDC)   // 512
#define NEG   0.2f

typedef __attribute__((ext_vector_type(16))) _Float16 v16h;
typedef __attribute__((ext_vector_type(8)))  float    v8f;

__device__ __forceinline__ float lrelu(float x) { return x > 0.f ? x : NEG * x; }

// monotone float <-> uint mapping so unsigned atomic max == float max
__device__ __forceinline__ unsigned f2ord(float f) {
  unsigned u = __float_as_uint(f);
  return (u & 0x80000000u) ? ~u : (u | 0x80000000u);
}
__device__ __forceinline__ float ord2f(unsigned u) {
  return __uint_as_float((u & 0x80000000u) ? (u & 0x7fffffffu) : ~u);
}
__device__ __forceinline__ void atomAddF(float* p, float v) {
  __hip_atomic_fetch_add(p, v, __ATOMIC_RELAXED, __HIP_MEMORY_SCOPE_AGENT);
}
__device__ __forceinline__ void atomMaxU(unsigned* p, unsigned v) {
  __hip_atomic_fetch_max(p, v, __ATOMIC_RELAXED, __HIP_MEMORY_SCOPE_AGENT);
}

// ---------------- WMMA GEMM: C[M,N] = A[M,K] @ B[K,N] + bias[N] ----------------
// Block = 8 waves, tile 128M x 32N. B tile staged in LDS (coalesced loads,
// stored in ISA 7.12.2 fragment layout, read back as ds_load_b128 broadcast).
// Each wave: 16 rows, 2 accumulators, 2 v_wmma per k-step, A fragment reused.
#define GBM 128
#define GBN 32

__global__ __launch_bounds__(256)
void gat_gemm_wmma(const float* __restrict__ A, const float* __restrict__ B,
                   const float* __restrict__ bias, float* __restrict__ C,
                   int M, int N, int K)
{
  __shared__ alignas(32) _Float16 lds_b[2 * 512]; // two 32x16 f16 fragments

  const int tid  = threadIdx.x;
  const int wv   = tid >> 5;
  const int lane = tid & 31;
  const int kh   = lane >> 4;   // K-half for A/B fragments, M-half for C
  const int lm   = lane & 15;

  const int n1 = blockIdx.x * GBN;
  const int m0 = blockIdx.y * GBM + wv * 16;
  const bool valid = (m0 + 16) <= M;     // uniform per wave
  const int arowi = valid ? (m0 + lm) : lm;

  v8f acc0, acc1;
  {
    float b0 = bias[n1 + lm];
    float b1 = bias[n1 + 16 + lm];
    #pragma unroll
    for (int r = 0; r < 8; ++r) { acc0[r] = b0; acc1[r] = b1; }
  }

  const float* arow = A + (size_t)arowi * K + (kh << 3);

  for (int k0 = 0; k0 < K; k0 += 32) {
    // ---- cooperative, coalesced B tile load: 32 K-rows x 32 N-cols ----
    #pragma unroll
    for (int i = 0; i < 4; ++i) {
      int e  = tid + i * 256;          // 0..1023
      int kk = e >> 5;                 // K row in tile
      int nn = e & 31;                 // N col in tile (consecutive per lane)
      float v = B[(size_t)(k0 + kk) * N + (n1 + nn)];
      // fragment layout: frag c = nn>>4, lane L = 16*((kk>>3)&1) + (nn&15), elem j = 8*(kk>>4) + (kk&7)
      lds_b[((nn >> 4) << 9) + (((((kk >> 3) & 1) << 4) + (nn & 15)) << 4) + (((kk >> 4) << 3) + (kk & 7))] =
          (_Float16)v;
    }
    __syncthreads();

    // ---- A 16x32 f16 fragment: elem j -> K = (j<8?0:16) + 8*kh + (j&7) ----
    const float4 p0 = *(const float4*)(arow + k0);
    const float4 p1 = *(const float4*)(arow + k0 + 4);
    const float4 p2 = *(const float4*)(arow + k0 + 16);
    const float4 p3 = *(const float4*)(arow + k0 + 20);
    v16h af;
    af[0]=(_Float16)p0.x;  af[1]=(_Float16)p0.y;  af[2]=(_Float16)p0.z;  af[3]=(_Float16)p0.w;
    af[4]=(_Float16)p1.x;  af[5]=(_Float16)p1.y;  af[6]=(_Float16)p1.z;  af[7]=(_Float16)p1.w;
    af[8]=(_Float16)p2.x;  af[9]=(_Float16)p2.y;  af[10]=(_Float16)p2.z; af[11]=(_Float16)p2.w;
    af[12]=(_Float16)p3.x; af[13]=(_Float16)p3.y; af[14]=(_Float16)p3.z; af[15]=(_Float16)p3.w;

    // ---- B fragments from LDS: 2x ds_load_b128 each, broadcast to all waves ----
    const v16h bf0 = *(const v16h*)(lds_b + (lane << 4));
    const v16h bf1 = *(const v16h*)(lds_b + 512 + (lane << 4));

    acc0 = __builtin_amdgcn_wmma_f32_16x16x32_f16(false, af, false, bf0,
                                                  (short)0, acc0, false, false);
    acc1 = __builtin_amdgcn_wmma_f32_16x16x32_f16(false, af, false, bf1,
                                                  (short)0, acc1, false, false);
    __syncthreads();   // protect LDS before next k-step's stores
  }

  if (valid) {
    float* c0 = C + (size_t)(m0 + (kh << 3)) * N + (n1 + lm);
    #pragma unroll
    for (int r = 0; r < 8; ++r) c0[(size_t)r * N] = acc0[r];
    float* c1 = c0 + 16;
    #pragma unroll
    for (int r = 0; r < 8; ++r) c1[(size_t)r * N] = acc1[r];
  }
}

// ---------------- Layer-1 attention kernels (4 heads, 128 ch) ----------------
__global__ __launch_bounds__(256)
void gat_logits1(const float* __restrict__ h1, const float* __restrict__ a_src,
                 const float* __restrict__ a_dst, float* __restrict__ als,
                 float* __restrict__ ald, int Nn)
{
  int t = blockIdx.x * blockDim.x + threadIdx.x;   // one thread per (node, head)
  if (t >= Nn * HEADS) return;
  int n = t >> 2, h = t & 3;
  const float4* hp = (const float4*)(h1 + (size_t)n * H1W + h * HIDC);
  const float4* as = (const float4*)(a_src + h * HIDC);
  const float4* ad = (const float4*)(a_dst + h * HIDC);
  float s = 0.f, d = 0.f;
  #pragma unroll 4
  for (int c = 0; c < HIDC / 4; ++c) {
    float4 v = hp[c], xx = as[c], yy = ad[c];
    s += v.x*xx.x + v.y*xx.y + v.z*xx.z + v.w*xx.w;
    d += v.x*yy.x + v.y*yy.y + v.z*yy.z + v.w*yy.w;
  }
  als[t] = s; ald[t] = d;
}

__global__ __launch_bounds__(256)
void gat_maxinit1(const float4* __restrict__ als, const float4* __restrict__ ald,
                  unsigned* __restrict__ emaxu, int Nn)
{
  int n = blockIdx.x * blockDim.x + threadIdx.x;
  if (n >= Nn) return;
  float4 a = als[n], b = ald[n];           // self-loop seeds the segment max
  uint4 u = make_uint4(f2ord(lrelu(a.x + b.x)), f2ord(lrelu(a.y + b.y)),
                       f2ord(lrelu(a.z + b.z)), f2ord(lrelu(a.w + b.w)));
  *(uint4*)(emaxu + (size_t)n * 4) = u;
}

__global__ __launch_bounds__(256)
void gat_edgemax1(const int* __restrict__ ei, int E, const float4* __restrict__ als,
                  const float4* __restrict__ ald, unsigned* __restrict__ emaxu)
{
  int e = blockIdx.x * blockDim.x + threadIdx.x;
  if (e >= E) return;
  int s = ei[e], d = ei[E + e];
  float4 a = als[s], b = ald[d];
  unsigned* p = emaxu + (size_t)d * 4;
  atomMaxU(p + 0, f2ord(lrelu(a.x + b.x)));
  atomMaxU(p + 1, f2ord(lrelu(a.y + b.y)));
  atomMaxU(p + 2, f2ord(lrelu(a.z + b.z)));
  atomMaxU(p + 3, f2ord(lrelu(a.w + b.w)));
}

// decode max; init denom and agg with the self-loop contribution
__global__ __launch_bounds__(128)
void gat_prep1(const float* __restrict__ h1, const float4* __restrict__ als,
               const float4* __restrict__ ald, const unsigned* __restrict__ emaxu,
               float4* __restrict__ emaxf, float4* __restrict__ denom,
               float* __restrict__ agg, int Nn)
{
  int n = blockIdx.x;
  if (n >= Nn) return;
  int t = threadIdx.x;
  float4 a = als[n], b = ald[n];
  float4 m = make_float4(ord2f(emaxu[(size_t)n*4+0]), ord2f(emaxu[(size_t)n*4+1]),
                         ord2f(emaxu[(size_t)n*4+2]), ord2f(emaxu[(size_t)n*4+3]));
  float4 w = make_float4(__expf(lrelu(a.x+b.x)-m.x), __expf(lrelu(a.y+b.y)-m.y),
                         __expf(lrelu(a.z+b.z)-m.z), __expf(lrelu(a.w+b.w)-m.w));
  if (t == 0) { emaxf[n] = m; denom[n] = w; }
  const float* hp = h1 + (size_t)n * H1W;
  float* ap = agg + (size_t)n * H1W;
  ap[t]       = hp[t]       * w.x;
  ap[t + 128] = hp[t + 128] * w.y;
  ap[t + 256] = hp[t + 256] * w.z;
  ap[t + 384] = hp[t + 384] * w.w;
}

// one wave per edge: 512-wide scatter-add with unnormalized softmax weights
__global__ __launch_bounds__(256)
void gat_edgeagg1(const int* __restrict__ ei, int E, const float* __restrict__ h1,
                  const float4* __restrict__ als, const float4* __restrict__ ald,
                  const float4* __restrict__ emaxf, float* __restrict__ denom,
                  float* __restrict__ agg)
{
  int e = (blockIdx.x * blockDim.x + threadIdx.x) >> 5;
  if (e >= E) return;
  int lane = threadIdx.x & 31;
  int s = ei[e], d = ei[E + e];
  float4 a = als[s], b = ald[d], m = emaxf[d];
  float w0 = __expf(lrelu(a.x + b.x) - m.x);
  float w1 = __expf(lrelu(a.y + b.y) - m.y);
  float w2 = __expf(lrelu(a.z + b.z) - m.z);
  float w3 = __expf(lrelu(a.w + b.w) - m.w);
  if (lane == 0) atomAddF(&denom[(size_t)d * 4 + 0], w0);
  if (lane == 1) atomAddF(&denom[(size_t)d * 4 + 1], w1);
  if (lane == 2) atomAddF(&denom[(size_t)d * 4 + 2], w2);
  if (lane == 3) atomAddF(&denom[(size_t)d * 4 + 3], w3);
  float wh = (lane < 8) ? w0 : (lane < 16) ? w1 : (lane < 24) ? w2 : w3;  // head = lane/8
  const float4* hs = (const float4*)(h1 + (size_t)s * H1W);
  float* ap = agg + (size_t)d * H1W;
  int base = lane * 16;
  #pragma unroll
  for (int i = 0; i < 4; ++i) {
    float4 v = hs[(base >> 2) + i];
    int idx = base + i * 4;
    atomAddF(&ap[idx + 0], v.x * wh);
    atomAddF(&ap[idx + 1], v.y * wh);
    atomAddF(&ap[idx + 2], v.z * wh);
    atomAddF(&ap[idx + 3], v.w * wh);
  }
}

__global__ __launch_bounds__(256)
void gat_final1(const float* __restrict__ agg, const float* __restrict__ denom,
                const float* __restrict__ b1, float* __restrict__ out, long total)
{
  long i = (long)blockIdx.x * blockDim.x + threadIdx.x;
  if (i >= total) return;
  int idx = (int)(i & (H1W - 1));
  long n = i >> 9;
  float den = denom[n * 4 + (idx >> 7)] + 1e-16f;
  float v = agg[i] / den + b1[idx];
  out[i] = v > 0.f ? v : (__expf(v) - 1.f);   // ELU
}

// ---------------- Layer-2 attention kernels (1 head, 64 ch) ----------------
__global__ __launch_bounds__(256)
void gat_logits2(const float* __restrict__ h2, const float* __restrict__ a_src,
                 const float* __restrict__ a_dst, float* __restrict__ als,
                 float* __restrict__ ald, int Nn)
{
  int n = blockIdx.x * blockDim.x + threadIdx.x;
  if (n >= Nn) return;
  const float4* hp = (const float4*)(h2 + (size_t)n * FOUT);
  const float4* as = (const float4*)a_src;
  const float4* ad = (const float4*)a_dst;
  float s = 0.f, d = 0.f;
  #pragma unroll
  for (int c = 0; c < FOUT / 4; ++c) {
    float4 v = hp[c], xx = as[c], yy = ad[c];
    s += v.x*xx.x + v.y*xx.y + v.z*xx.z + v.w*xx.w;
    d += v.x*yy.x + v.y*yy.y + v.z*yy.z + v.w*yy.w;
  }
  als[n] = s; ald[n] = d;
}

__global__ __launch_bounds__(256)
void gat_maxinit2(const float* __restrict__ als, const float* __restrict__ ald,
                  unsigned* __restrict__ emaxu, int Nn)
{
  int n = blockIdx.x * blockDim.x + threadIdx.x;
  if (n >= Nn) return;
  emaxu[n] = f2ord(lrelu(als[n] + ald[n]));
}

__global__ __launch_bounds__(256)
void gat_edgemax2(const int* __restrict__ ei, int E, const float* __restrict__ als,
                  const float* __restrict__ ald, unsigned* __restrict__ emaxu)
{
  int e = blockIdx.x * blockDim.x + threadIdx.x;
  if (e >= E) return;
  int s = ei[e], d = ei[E + e];
  atomMaxU(&emaxu[d], f2ord(lrelu(als[s] + ald[d])));
}

__global__ __launch_bounds__(64)
void gat_prep2(const float* __restrict__ h2, const float* __restrict__ als,
               const float* __restrict__ ald, const unsigned* __restrict__ emaxu,
               float* __restrict__ emaxf, float* __restrict__ denom,
               float* __restrict__ agg, int Nn)
{
  int n = blockIdx.x;
  if (n >= Nn) return;
  int t = threadIdx.x;
  float m = ord2f(emaxu[n]);
  float w = __expf(lrelu(als[n] + ald[n]) - m);
  if (t == 0) { emaxf[n] = m; denom[n] = w; }
  agg[(size_t)n * FOUT + t] = h2[(size_t)n * FOUT + t] * w;
}

__global__ __launch_bounds__(256)
void gat_edgeagg2(const int* __restrict__ ei, int E, const float* __restrict__ h2,
                  const float* __restrict__ als, const float* __restrict__ ald,
                  const float* __restrict__ emaxf, float* __restrict__ denom,
                  float* __restrict__ agg)
{
  int e = (blockIdx.x * blockDim.x + threadIdx.x) >> 5;
  if (e >= E) return;
  int lane = threadIdx.x & 31;
  int s = ei[e], d = ei[E + e];
  float w = __expf(lrelu(als[s] + ald[d]) - emaxf[d]);
  if (lane == 0) atomAddF(&denom[d], w);
  const float2 v = *(const float2*)(h2 + (size_t)s * FOUT + lane * 2);
  float* ap = agg + (size_t)d * FOUT + lane * 2;
  atomAddF(&ap[0], v.x * w);
  atomAddF(&ap[1], v.y * w);
}

__global__ __launch_bounds__(256)
void gat_final2(const float* __restrict__ agg, const float* __restrict__ denom,
                const float* __restrict__ b2, float* __restrict__ out, int Nn)
{
  long i = (long)blockIdx.x * blockDim.x + threadIdx.x;
  if (i >= (long)Nn * FOUT) return;
  int c = (int)(i & (FOUT - 1));
  long n = i >> 6;
  out[i] = agg[i] / (denom[n] + 1e-16f) + b2[c];
}

// ---------------- host launcher ----------------
extern "C" void kernel_launch(void* const* d_in, const int* in_sizes, int n_in,
                              void* d_out, int out_size, void* d_ws, size_t ws_size,
                              hipStream_t stream)
{
  const float* x      = (const float*)d_in[0];
  const int*   ei     = (const int*)d_in[1];     // [2, E] flat: row0=src, row1=dst
  const float* W1     = (const float*)d_in[2];
  const float* a_src1 = (const float*)d_in[3];
  const float* a_dst1 = (const float*)d_in[4];
  const float* b1     = (const float*)d_in[5];
  const float* W2     = (const float*)d_in[6];
  const float* a_src2 = (const float*)d_in[7];
  const float* a_dst2 = (const float*)d_in[8];
  const float* b2     = (const float*)d_in[9];

  const int Nn = in_sizes[0] / FIN;
  const int E  = in_sizes[1] / 2;

  // workspace carve-out (256B aligned slices)
  char* w = (char*)d_ws;
  size_t off = 0;
  auto carve = [&](size_t bytes) -> void* {
    void* p = w + off;
    off = (off + bytes + 255) & ~(size_t)255;
    return p;
  };
  float*    h1     = (float*)   carve((size_t)Nn * H1W * 4);  // also reused as ELU out
  float*    agg1   = (float*)   carve((size_t)Nn * H1W * 4);  // layer-2 bufs reuse this
  float*    als1   = (float*)   carve((size_t)Nn * HEADS * 4);
  float*    ald1   = (float*)   carve((size_t)Nn * HEADS * 4);
  unsigned* emax1u = (unsigned*)carve((size_t)Nn * HEADS * 4);
  float*    emaxf1 = (float*)   carve((size_t)Nn * HEADS * 4);
  float*    denom1 = (float*)   carve((size_t)Nn * HEADS * 4);
  float*    als2   = (float*)   carve((size_t)Nn * 4);
  float*    ald2   = (float*)   carve((size_t)Nn * 4);
  unsigned* emax2u = (unsigned*)carve((size_t)Nn * 4);
  float*    emaxf2 = (float*)   carve((size_t)Nn * 4);
  float*    denom2 = (float*)   carve((size_t)Nn * 4);
  // agg1 region is dead after gat_final1; place layer-2 feature bufs inside it
  float* h2   = agg1;
  float* agg2 = agg1 + (size_t)Nn * FOUT;

  auto cdiv = [](long a, long b) { return (int)((a + b - 1) / b); };

  // ---- layer 1 ----
  {
    dim3 g(H1W / GBN, cdiv(Nn, GBM));
    gat_gemm_wmma<<<g, 256, 0, stream>>>(x, W1, b1, h1, Nn, H1W, FIN);
  }
  gat_logits1 <<<cdiv((long)Nn * HEADS, 256), 256, 0, stream>>>(h1, a_src1, a_dst1, als1, ald1, Nn);
  gat_maxinit1<<<cdiv(Nn, 256), 256, 0, stream>>>((const float4*)als1, (const float4*)ald1, emax1u, Nn);
  gat_edgemax1<<<cdiv(E, 256), 256, 0, stream>>>(ei, E, (const float4*)als1, (const float4*)ald1, emax1u);
  gat_prep1   <<<Nn, 128, 0, stream>>>(h1, (const float4*)als1, (const float4*)ald1, emax1u,
                                       (float4*)emaxf1, (float4*)denom1, agg1, Nn);
  gat_edgeagg1<<<cdiv(E, 8), 256, 0, stream>>>(ei, E, h1, (const float4*)als1,
                                               (const float4*)ald1, (const float4*)emaxf1,
                                               denom1, agg1);
  gat_final1  <<<cdiv((long)Nn * H1W, 256), 256, 0, stream>>>(agg1, denom1, b1, h1, (long)Nn * H1W);

  // ---- layer 2 ----
  {
    dim3 g(FOUT / GBN, cdiv(Nn, GBM));
    gat_gemm_wmma<<<g, 256, 0, stream>>>(h1, W2, b2, h2, Nn, FOUT, H1W);
  }
  gat_logits2 <<<cdiv(Nn, 256), 256, 0, stream>>>(h2, a_src2, a_dst2, als2, ald2, Nn);
  gat_maxinit2<<<cdiv(Nn, 256), 256, 0, stream>>>(als2, ald2, emax2u, Nn);
  gat_edgemax2<<<cdiv(E, 256), 256, 0, stream>>>(ei, E, als2, ald2, emax2u);
  gat_prep2   <<<Nn, 64, 0, stream>>>(h2, als2, ald2, emax2u, emaxf2, denom2, agg2, Nn);
  gat_edgeagg2<<<cdiv(E, 8), 256, 0, stream>>>(ei, E, h2, als2, ald2, emaxf2, denom2, agg2);
  gat_final2  <<<cdiv((long)Nn * FOUT, 256), 256, 0, stream>>>(agg2, denom2, b2, (float*)d_out, Nn);
}
